// VariationalAutoEncoder_9174050144471
// MI455X (gfx1250) — matrix operand order repaired and checked by reading, not compiled, not run on hardware
//
#include <hip/hip_runtime.h>

typedef __attribute__((ext_vector_type(16))) __bf16 v16bf;
typedef __attribute__((ext_vector_type(4)))  __bf16 v4bf;
typedef __attribute__((ext_vector_type(8)))  float  v8f;

#define TILE_M 128
#define TILE_F 128
#define TILE_K 32
#define LDSS   40   // padded row stride in halfwords (80 bytes, 16B aligned)

// ---------------------------------------------------------------------------
// fp32 -> bf16 via hardware convert (fptrunc to bfloat = RNE)
__device__ __forceinline__ unsigned short f32_to_bf16(float f) {
    __bf16 h = (__bf16)f;
    return __builtin_bit_cast(unsigned short, h);
}

// out[c*rows + r] = bf16(in[r*cols + c])   (pre-transpose B operand)
__global__ __launch_bounds__(256)
void transpose_to_bf16(const float* __restrict__ in, unsigned short* __restrict__ out,
                       int rows, int cols) {
    long long i = (long long)blockIdx.x * 256 + threadIdx.x;
    long long total = (long long)rows * cols;
    if (i >= total) return;
    int r = (int)(i / cols), c = (int)(i % cols);
    out[(long long)c * rows + r] = f32_to_bf16(in[i]);
}

// ---------------------------------------------------------------------------
// C[M x F] = A[M x K] (fp32) * Bt[F x K] (bf16, pre-transposed)  [+bias][+relu]
// mode: 0 = none, 1 = +bias, 2 = +bias, relu
__global__ __launch_bounds__(256)
void wmma_gemm_bf16(const float* __restrict__ A, const unsigned short* __restrict__ Bt,
                    const float* __restrict__ bias, float* __restrict__ C,
                    int M, int K, int F, int mode) {
    __shared__ __attribute__((aligned(16))) unsigned short lA[TILE_M * LDSS];
    __shared__ __attribute__((aligned(16))) unsigned short lB[TILE_F * LDSS];

    const int tid  = threadIdx.x;
    const int wave = tid >> 5;
    const int lane = tid & 31;
    const int lm   = lane & 15;
    const int hi   = lane >> 4;          // 0 or 1
    const int m0   = blockIdx.x * TILE_M;
    const int f0   = blockIdx.y * TILE_F;

    v8f acc[8];
    #pragma unroll
    for (int i = 0; i < 8; ++i)
        #pragma unroll
        for (int j = 0; j < 8; ++j) acc[i][j] = 0.0f;

    union Frag { v16bf v; uint4 q[2]; };

    for (int k0 = 0; k0 < K; k0 += TILE_K) {
        // A tile: 128 rows x 32 k, fp32 -> bf16 into LDS (float4 reads, b64 stores)
        #pragma unroll
        for (int i = tid; i < (TILE_M * TILE_K) / 4; i += 256) {
            int r = i >> 3, c4 = i & 7;                     // 8 float4 per row
            float4 v = *(const float4*)(A + (long long)(m0 + r) * K + k0 + c4 * 4);
            v4bf hv;
            hv.x = (__bf16)v.x;
            hv.y = (__bf16)v.y;
            hv.z = (__bf16)v.z;
            hv.w = (__bf16)v.w;
            *(uint2*)&lA[r * LDSS + c4 * 4] = __builtin_bit_cast(uint2, hv);
        }
        // B tile: 128 out-cols x 32 k, already bf16 (uint4 reads, b128 stores)
        #pragma unroll
        for (int i = tid; i < (TILE_F * TILE_K) / 8; i += 256) {
            int r = i >> 2, c8 = i & 3;                     // 4 uint4 per row
            uint4 v = *(const uint4*)(Bt + (long long)(f0 + r) * K + k0 + c8 * 8);
            *(uint4*)&lB[r * LDSS + c8 * 8] = v;
        }
        __syncthreads();

        // A fragment (ISA 16-bit A 16x32): v0-3 K=8*hi+{0..7}, v4-7 K=16+8*hi+{0..7}
        Frag fa;
        {
            const unsigned short* rp = &lA[(wave * 16 + lm) * LDSS];
            fa.q[0] = *(const uint4*)(rp + 8 * hi);
            fa.q[1] = *(const uint4*)(rp + 16 + 8 * hi);
        }
        // preload ALL B fragments, then issue WMMAs back-to-back
        Frag fb[8];
        #pragma unroll
        for (int ct = 0; ct < 8; ++ct) {
            // B fragment (ISA B 32x16): v0-3 K=8*hi+{0..7}, v4-7 K=16+8*hi+{0..7}
            const unsigned short* rp = &lB[(ct * 16 + lm) * LDSS];
            fb[ct].q[0] = *(const uint4*)(rp + 8 * hi);
            fb[ct].q[1] = *(const uint4*)(rp + 16 + 8 * hi);
        }
        #pragma unroll
        for (int ct = 0; ct < 8; ++ct) {
            acc[ct] = __builtin_amdgcn_wmma_f32_16x16x32_bf16(
                false, fa.v, false, fb[ct].v, (short)0, acc[ct], false, false);
        }
        __syncthreads();
    }

    // C layout: VGPR r, lane L -> row = wave*16 + r + 8*hi, col = ct*16 + (L&15)
    const int rowb = m0 + wave * 16 + 8 * hi;
    #pragma unroll
    for (int ct = 0; ct < 8; ++ct) {
        int col = f0 + ct * 16 + lm;
        float bv = (mode != 0) ? bias[col] : 0.0f;
        #pragma unroll
        for (int r = 0; r < 8; ++r) {
            float v = acc[ct][r] + bv;
            if (mode == 2) v = fmaxf(v, 0.0f);
            C[(long long)(rowb + r) * F + col] = v;
        }
    }
}

// ---------------------------------------------------------------------------
// deterministic segment sum: idx sorted, one block per segment, binary search
__global__ __launch_bounds__(256)
void segment_sum_kernel(const float* __restrict__ h, const int* __restrict__ idx,
                        float* __restrict__ g) {
    int b = blockIdx.x, f = threadIdx.x;
    __shared__ int lo_s, hi_s;
    if (f == 0) {
        int lo = 0, hi = 8192;
        while (lo < hi) { int mid = (lo + hi) >> 1; if (idx[mid] < b) lo = mid + 1; else hi = mid; }
        lo_s = lo;
        int lo2 = lo, hi2 = 8192;
        while (lo2 < hi2) { int mid = (lo2 + hi2) >> 1; if (idx[mid] <= b) lo2 = mid + 1; else hi2 = mid; }
        hi_s = lo2;
    }
    __syncthreads();
    float s = 0.0f;
    for (int n = lo_s; n < hi_s; ++n) s += h[(long long)n * 256 + f];
    g[b * 256 + f] = s;
}

// ---------------------------------------------------------------------------
// g -> fc -> mu/logvar -> z (one block per batch row)
__global__ __launch_bounds__(256)
void head_kernel(const float* __restrict__ g,
                 const float* __restrict__ Wfc, const float* __restrict__ bfc,
                 const float* __restrict__ Wmu, const float* __restrict__ bmu,
                 const float* __restrict__ Wlv, const float* __restrict__ blv,
                 const float* __restrict__ eps,
                 float* __restrict__ mu, float* __restrict__ logvar, float* __restrict__ z) {
    int b = blockIdx.x, o = threadIdx.x;
    __shared__ float row[256];
    __shared__ float gf[256];
    row[o] = g[b * 256 + o];
    __syncthreads();
    float s = bfc[o];
    for (int k = 0; k < 256; ++k) s += row[k] * Wfc[k * 256 + o];
    gf[o] = s;
    __syncthreads();
    if (o < 64) {
        float sm = bmu[o], sl = blv[o];
        for (int k = 0; k < 256; ++k) {
            float gv = gf[k];
            sm += gv * Wmu[k * 64 + o];
            sl += gv * Wlv[k * 64 + o];
        }
        mu[b * 64 + o] = sm;
        logvar[b * 64 + o] = sl;
        z[b * 64 + o] = sm + eps[b * 64 + o] * __expf(0.5f * sl);
    }
}

// ---------------------------------------------------------------------------
// out = LayerNorm(relu(in @ W + b)) * gamma + beta, one block per row
__global__ __launch_bounds__(256)
void dec_layer_kernel(const float* __restrict__ in, const float* __restrict__ W,
                      const float* __restrict__ bias, const float* __restrict__ gamma,
                      const float* __restrict__ beta, float* __restrict__ out,
                      int din, int dout) {
    __shared__ float s_in[512];
    __shared__ float s_h[768];
    __shared__ float s_red[256];
    int b = blockIdx.x, t = threadIdx.x;
    for (int i = t; i < din; i += 256) s_in[i] = in[(long long)b * din + i];
    __syncthreads();
    for (int o = t; o < dout; o += 256) {
        float s = bias[o];
        for (int k = 0; k < din; ++k) s += s_in[k] * W[(long long)k * dout + o];
        s_h[o] = fmaxf(s, 0.0f);
    }
    __syncthreads();
    float ls = 0.0f, ls2 = 0.0f;
    for (int o = t; o < dout; o += 256) { float v = s_h[o]; ls += v; ls2 += v * v; }
    s_red[t] = ls;
    __syncthreads();
    for (int off = 128; off > 0; off >>= 1) { if (t < off) s_red[t] += s_red[t + off]; __syncthreads(); }
    float mean = s_red[0] / (float)dout;
    __syncthreads();
    s_red[t] = ls2;
    __syncthreads();
    for (int off = 128; off > 0; off >>= 1) { if (t < off) s_red[t] += s_red[t + off]; __syncthreads(); }
    float var = s_red[0] / (float)dout - mean * mean;
    float inv = rsqrtf(var + 1e-5f);
    for (int o = t; o < dout; o += 256)
        out[(long long)b * dout + o] = (s_h[o] - mean) * inv * gamma[o] + beta[o];
}

// ---------------------------------------------------------------------------
// per-batch weighted BCE over symmetrized upper triangle -> partial[b]
__global__ __launch_bounds__(256)
void recon_kernel(const float* __restrict__ logits, const float* __restrict__ y,
                  float* __restrict__ partial) {
    int b = blockIdx.x, t = threadIdx.x;
    __shared__ float s_red[256];
    const float POSW_M1 = 1.5f; // POS_W - 1 = 1/0.4 - 1
    float acc = 0.0f;
    for (int p = t; p < 4096; p += 256) {
        int i = p >> 6, j = p & 63;
        if (j < i) continue;
        float lx = 0.5f * (logits[(long long)b * 4096 + i * 64 + j] +
                           logits[(long long)b * 4096 + j * 64 + i]);
        float ly = y[(long long)b * 4096 + i * 64 + j];
        // softplus(-lx), numerically stable
        float sp = fmaxf(-lx, 0.0f) + log1pf(expf(-fabsf(lx)));
        acc += (1.0f - ly) * lx + (1.0f + POSW_M1 * ly) * sp;
    }
    s_red[t] = acc;
    __syncthreads();
    for (int off = 128; off > 0; off >>= 1) { if (t < off) s_red[t] += s_red[t + off]; __syncthreads(); }
    if (t == 0) partial[b] = s_red[0];
}

// ---------------------------------------------------------------------------
// sum partials, compute KLD, final loss
__global__ __launch_bounds__(256)
void finalize_kernel(const float* __restrict__ partial, const float* __restrict__ mu,
                     const float* __restrict__ logvar, float* __restrict__ out) {
    __shared__ float s_red[256];
    int t = threadIdx.x;
    float r = partial[t];
    float k = 0.0f;
    for (int i = t; i < 256 * 64; i += 256) {
        float m = mu[i], lv = logvar[i];
        k += 1.0f + lv - m * m - expf(lv);
    }
    s_red[t] = r;
    __syncthreads();
    for (int off = 128; off > 0; off >>= 1) { if (t < off) s_red[t] += s_red[t + off]; __syncthreads(); }
    float recon = s_red[0];
    __syncthreads();
    s_red[t] = k;
    __syncthreads();
    for (int off = 128; off > 0; off >>= 1) { if (t < off) s_red[t] += s_red[t + off]; __syncthreads(); }
    if (t == 0) {
        float kld = -0.5f * s_red[0];
        out[0] = recon + 0.05f * kld;
        out[1] = recon;
        out[2] = kld;
    }
}

// ---------------------------------------------------------------------------
extern "C" void kernel_launch(void* const* d_in, const int* in_sizes, int n_in,
                              void* d_out, int out_size, void* d_ws, size_t ws_size,
                              hipStream_t stream) {
    (void)in_sizes; (void)n_in; (void)out_size; (void)ws_size;
    const float* adj  = (const float*)d_in[0];
    const float* x    = (const float*)d_in[1];
    const float* y    = (const float*)d_in[2];
    const float* eps  = (const float*)d_in[3];
    const float* W1_0 = (const float*)d_in[4];
    const float* b1_0 = (const float*)d_in[5];
    const float* W2_0 = (const float*)d_in[6];
    const float* b2_0 = (const float*)d_in[7];
    const float* W1_1 = (const float*)d_in[8];
    const float* b1_1 = (const float*)d_in[9];
    const float* W2_1 = (const float*)d_in[10];
    const float* b2_1 = (const float*)d_in[11];
    const float* Wfc  = (const float*)d_in[12];
    const float* bfc  = (const float*)d_in[13];
    const float* Wmu  = (const float*)d_in[14];
    const float* bmu  = (const float*)d_in[15];
    const float* Wlv  = (const float*)d_in[16];
    const float* blv  = (const float*)d_in[17];
    const float* dW0  = (const float*)d_in[18];
    const float* db0  = (const float*)d_in[19];
    const float* dg0  = (const float*)d_in[20];
    const float* dbe0 = (const float*)d_in[21];
    const float* dW1  = (const float*)d_in[22];
    const float* db1  = (const float*)d_in[23];
    const float* dg1  = (const float*)d_in[24];
    const float* dbe1 = (const float*)d_in[25];
    const float* dW2  = (const float*)d_in[26];
    const float* db2  = (const float*)d_in[27];
    const float* dg2  = (const float*)d_in[28];
    const float* dbe2 = (const float*)d_in[29];
    const float* Wpj  = (const float*)d_in[30];
    const float* bpj  = (const float*)d_in[31];
    const int*   idx  = (const int*)d_in[32];

    float* ws = (float*)d_ws;
    const long long H = 8192LL * 256;            // 2M floats
    float* h0     = ws;
    float* h1     = ws + H;
    unsigned short* Bt = (unsigned short*)(ws + 2 * H);   // up to 4096*768 bf16 (fits 2M floats)
    float* g      = ws + 3 * H;
    float* mu     = g + 65536;
    float* lv     = mu + 16384;
    float* z      = lv + 16384;
    float* hd1    = z + 16384;
    float* hd2    = hd1 + 65536;
    float* hd3    = hd2 + 131072;
    float* logits = hd3 + 196608;
    float* part   = logits + 1048576;

    auto T = [&](const float* in, int rows, int cols) {
        long long tot = (long long)rows * cols;
        int blocks = (int)((tot + 255) / 256);
        transpose_to_bf16<<<dim3(blocks), dim3(256), 0, stream>>>(in, Bt, rows, cols);
    };
    auto G = [&](const float* A, const float* bias, float* C, int M, int K, int F, int mode) {
        wmma_gemm_bf16<<<dim3(M / 128, F / 128), dim3(256), 0, stream>>>(A, Bt, bias, C, M, K, F, mode);
    };

    // ---- encoder layer 0 ----
    T(x, 8192, 128);                         // Bt = x^T (bf16)
    G(adj, nullptr, h0, 8192, 8192, 128, 0); // h0 = adj @ x
    T(W1_0, 128, 256);
    G(h0, b1_0, h1, 8192, 128, 256, 2);      // h1 = relu(h0 @ W1 + b1)
    T(W2_0, 256, 256);
    G(h1, b2_0, h0, 8192, 256, 256, 2);      // h0 = relu(h1 @ W2 + b2)

    // ---- encoder layer 1 ----
    T(h0, 8192, 256);                        // Bt = h0^T
    G(adj, nullptr, h1, 8192, 8192, 256, 0); // h1 = adj @ h0
    T(W1_1, 256, 256);
    G(h1, b1_1, h0, 8192, 256, 256, 2);
    T(W2_1, 256, 256);
    G(h0, b2_1, h1, 8192, 256, 256, 2);      // h1 = final node features

    // ---- pooling + variational head ----
    segment_sum_kernel<<<dim3(256), dim3(256), 0, stream>>>(h1, idx, g);
    head_kernel<<<dim3(256), dim3(256), 0, stream>>>(g, Wfc, bfc, Wmu, bmu, Wlv, blv,
                                                     eps, mu, lv, z);

    // ---- decoder ----
    dec_layer_kernel<<<dim3(256), dim3(256), 0, stream>>>(z,   dW0, db0, dg0, dbe0, hd1,  64, 256);
    dec_layer_kernel<<<dim3(256), dim3(256), 0, stream>>>(hd1, dW1, db1, dg1, dbe1, hd2, 256, 512);
    dec_layer_kernel<<<dim3(256), dim3(256), 0, stream>>>(hd2, dW2, db2, dg2, dbe2, hd3, 512, 768);

    // ---- projection to logits (WMMA) ----
    T(Wpj, 768, 4096);
    G(hd3, bpj, logits, 256, 768, 4096, 1);

    // ---- losses ----
    recon_kernel<<<dim3(256), dim3(256), 0, stream>>>(logits, y, part);
    finalize_kernel<<<dim3(1), dim3(256), 0, stream>>>(part, mu, lv, (float*)d_out);
}
